// AttnBlock_54065048322916
// MI455X (gfx1250) — matrix operand order repaired and checked by reading, not compiled, not run on hardware
//
#include <hip/hip_runtime.h>
#include <hip/hip_bf16.h>

typedef __attribute__((ext_vector_type(16))) _Float16 v16h;
typedef __attribute__((ext_vector_type(8)))  _Float16 v8h;
typedef __attribute__((ext_vector_type(8)))  float    v8f;
typedef __attribute__((ext_vector_type(4)))  float    v4f;
typedef __attribute__((ext_vector_type(4)))  int      v4i;

#define BATCH 32
#define CCH   512
#define NTOK  1024
#define NGRP  16
#define CG    32      // CCH / NGRP
#define LDT   40      // padded LDS row stride (halves)

#if defined(__AMDGCN__) && __has_builtin(__builtin_amdgcn_global_load_async_to_lds_b128) && __has_builtin(__builtin_amdgcn_s_wait_asynccnt)
#define HAVE_ASYNC_LDS 1
#define AS1 __attribute__((address_space(1)))
#define AS3 __attribute__((address_space(3)))
#else
#define HAVE_ASYNC_LDS 0
#endif

// ---------------- weights fp32 -> fp16 (packed wq|wk|wv|wp) ----------------
__global__ __launch_bounds__(256) void cvt_weights(
    const float* __restrict__ wq, const float* __restrict__ wk,
    const float* __restrict__ wv, const float* __restrict__ wp,
    _Float16* __restrict__ dst)
{
  int i = blockIdx.x * 256 + threadIdx.x;          // 4 * 512 * 512 total
  int w = i >> 18;
  int off = i & 0x3FFFF;
  const float* s = (w == 0) ? wq : (w == 1) ? wk : (w == 2) ? wv : wp;
  dst[i] = (_Float16)s[off];
}

// ---------------- GroupNorm: x(b,c,n) f32 -> hnT(b,n,c) f16 ----------------
__global__ __launch_bounds__(256) void groupnorm_to_hnT(
    const float* __restrict__ x, const float* __restrict__ gamma,
    const float* __restrict__ beta, _Float16* __restrict__ hnT)
{
  const int g = blockIdx.x, b = blockIdx.y, tid = threadIdx.x;
  const float* base = x + ((size_t)b * CCH + (size_t)g * CG) * NTOK;
  const int NE = CG * NTOK;                        // 32768
  float s1 = 0.f, s2 = 0.f;
  const v4f* b4 = (const v4f*)base;
  for (int i = tid; i < NE / 4; i += 256) {
    v4f v = b4[i];
    s1 += v.x + v.y + v.z + v.w;
    s2 += v.x * v.x + v.y * v.y + v.z * v.z + v.w * v.w;
  }
  __shared__ float r1[256], r2[256];
  r1[tid] = s1; r2[tid] = s2;
  __syncthreads();
  for (int s = 128; s > 0; s >>= 1) {
    if (tid < s) { r1[tid] += r1[tid + s]; r2[tid] += r2[tid + s]; }
    __syncthreads();
  }
  const float mean = r1[0] * (1.f / NE);
  const float var  = r2[0] * (1.f / NE) - mean * mean;
  const float inv  = rsqrtf(var + 1e-6f);
  for (int i = tid; i < NE; i += 256) {
    int cl = i >> 10, n = i & (NTOK - 1);
    int c  = g * CG + cl;
    float v = (base[i] - mean) * inv * gamma[c] + beta[c];
    hnT[((size_t)b * NTOK + n) * CCH + c] = (_Float16)v;
  }
}

// ---- stage one 128x32 A tile chunk + 128x32 B tile chunk (per thread: 2+2 b128)
__device__ __forceinline__ void stage_tiles(
    const _Float16* __restrict__ A, const _Float16* __restrict__ Bm, int K,
    int m0, int n0, int k0, _Float16* la, _Float16* lb, int tid)
{
  const int row = tid >> 1;             // 0..127
  const int cc  = (tid & 1) * 16;       // 0 / 16 halves
  const _Float16* sa = A  + (size_t)(m0 + row) * K + k0 + cc;
  const _Float16* sb = Bm + (size_t)(n0 + row) * K + k0 + cc;
  _Float16* da = la + row * LDT + cc;
  _Float16* db = lb + row * LDT + cc;
#if HAVE_ASYNC_LDS
  __builtin_amdgcn_global_load_async_to_lds_b128((AS1 v4i*)sa,       (AS3 v4i*)da,       0, 0);
  __builtin_amdgcn_global_load_async_to_lds_b128((AS1 v4i*)(sa + 8), (AS3 v4i*)(da + 8), 0, 0);
  __builtin_amdgcn_global_load_async_to_lds_b128((AS1 v4i*)sb,       (AS3 v4i*)db,       0, 0);
  __builtin_amdgcn_global_load_async_to_lds_b128((AS1 v4i*)(sb + 8), (AS3 v4i*)(db + 8), 0, 0);
#else
  *(v8h*)da       = *(const v8h*)sa;
  *(v8h*)(da + 8) = *(const v8h*)(sa + 8);
  *(v8h*)db       = *(const v8h*)sb;
  *(v8h*)(db + 8) = *(const v8h*)(sb + 8);
#endif
}

__device__ __forceinline__ void stage_wait()
{
#if HAVE_ASYNC_LDS
  __builtin_amdgcn_s_wait_asynccnt(0);
#endif
}

// ---------------- Generic WMMA GEMM: D = A(MxK) * B(NCxK)^T ----------------
// 128x128 block tile, K-step 32, double-buffered LDS, 8 wave32 waves (4x2),
// wave tile 32x64 = eight 16x16 f32 accumulators -> 8 v_wmma per k-step/wave.
// MODE 0: outH f16 natural, scaled          (scores)
// MODE 1: outH f16 TRANSPOSED [+bias]       (q^T, k^T, out_h^T)
// MODE 2: outH f16 natural + bias           (v)
// MODE 3: outF f32 natural + bias + residual (projection + skip)
template <int MODE>
__global__ __launch_bounds__(256) void gemm16(
    const _Float16* __restrict__ A,  size_t aStride,
    const _Float16* __restrict__ Bm, size_t bStride,
    _Float16* __restrict__ outH, float* __restrict__ outF, size_t oStride,
    const float* __restrict__ bias,
    const float* __restrict__ resid, size_t rStride,
    int M, int K, float scale)
{
  __shared__ _Float16 ldsA[2][128 * LDT];
  __shared__ _Float16 ldsB[2][128 * LDT];

  const int b = blockIdx.z;
  A  += (size_t)b * aStride;
  Bm += (size_t)b * bStride;
  if (MODE == 3) { outF += (size_t)b * oStride; resid += (size_t)b * rStride; }
  else           { outH += (size_t)b * oStride; }

  const int n0 = blockIdx.x * 128;
  const int m0 = blockIdx.y * 128;

  const int tid  = threadIdx.x;
  const int wave = tid >> 5;
  const int lane = tid & 31;
  const int hi   = lane >> 4;     // half-wave select
  const int ln   = lane & 15;
  const int wm   = wave & 3;      // 4 waves along M (32 rows each)
  const int wn   = wave >> 2;     // 2 waves along N (64 cols each)

  v8f acc[2][4];
  for (int i = 0; i < 2; ++i)
    for (int j = 0; j < 4; ++j)
      for (int r = 0; r < 8; ++r) acc[i][j][r] = 0.f;

  const int steps = K >> 5;
  stage_tiles(A, Bm, K, m0, n0, 0, ldsA[0], ldsB[0], tid);
  stage_wait();
  __syncthreads();

  for (int kt = 0; kt < steps; ++kt) {
    const int cur = kt & 1;
    if (kt + 1 < steps)
      stage_tiles(A, Bm, K, m0, n0, (kt + 1) << 5, ldsA[cur ^ 1], ldsB[cur ^ 1], tid);

    // fragments (ISA 16-bit A 16x32 / B 32x16 layouts), contiguous b128 LDS reads
    v16h af[2], bf[4];
    const int koffA = hi * 8;
    for (int tm = 0; tm < 2; ++tm) {
      int m = wm * 32 + tm * 16 + ln;
      v8h a0 = *(v8h*)&ldsA[cur][m * LDT + koffA];
      v8h a1 = *(v8h*)&ldsA[cur][m * LDT + koffA + 16];
      af[tm] = __builtin_shufflevector(a0, a1, 0,1,2,3,4,5,6,7,8,9,10,11,12,13,14,15);
    }
    const int koffB = hi * 8;
    for (int tn = 0; tn < 4; ++tn) {
      int n = wn * 64 + tn * 16 + ln;
      v8h b0 = *(v8h*)&ldsB[cur][n * LDT + koffB];
      v8h b1 = *(v8h*)&ldsB[cur][n * LDT + koffB + 16];
      bf[tn] = __builtin_shufflevector(b0, b1, 0,1,2,3,4,5,6,7,8,9,10,11,12,13,14,15);
    }

    for (int tm = 0; tm < 2; ++tm)
      for (int tn = 0; tn < 4; ++tn)
        acc[tm][tn] = __builtin_amdgcn_wmma_f32_16x16x32_f16(
            false, af[tm], false, bf[tn], (short)0, acc[tm][tn], false, false);

    if (kt + 1 < steps) {
      stage_wait();
      __syncthreads();
    }
  }

  // ---- epilogues (branch-free per instantiation) ----
  for (int tm = 0; tm < 2; ++tm)
    for (int tn = 0; tn < 4; ++tn) {
      const int rowb = m0 + wm * 32 + tm * 16 + hi * 8;   // C/D: VGPR r -> M=r(+8)
      const int col  = n0 + wn * 64 + tn * 16 + ln;       // lanes -> N
      v8f a = acc[tm][tn];
      if (MODE == 0) {
        for (int r = 0; r < 8; ++r)
          outH[(size_t)(rowb + r) * NTOK + col] = (_Float16)(a[r] * scale);
      } else if (MODE == 1) {
        for (int r = 0; r < 8; ++r) {
          int row = rowb + r;
          float v = a[r] + (bias ? bias[row] : 0.f);
          outH[(size_t)col * M + row] = (_Float16)v;       // transposed store
        }
      } else if (MODE == 2) {
        for (int r = 0; r < 8; ++r) {
          int row = rowb + r;
          outH[(size_t)row * NTOK + col] = (_Float16)(a[r] + bias[row]);
        }
      } else {
        for (int r = 0; r < 8; ++r) {
          int row = rowb + r;
          size_t idx = (size_t)row * NTOK + col;
          outF[idx] = a[r] + bias[row] + resid[idx];
        }
      }
    }
}

// ---------------- Row softmax over m, in place on f16 scores ----------------
__global__ __launch_bounds__(256) void softmax_rows(_Float16* __restrict__ s)
{
  const int n = blockIdx.x, b = blockIdx.y, tid = threadIdx.x;
  _Float16* row = s + ((size_t)b * NTOK + n) * NTOK;
  __shared__ float red[256];
  float vals[4];
  float m = -3.0e38f;
  for (int i = 0; i < 4; ++i) {
    vals[i] = (float)row[tid + i * 256];
    m = fmaxf(m, vals[i]);
  }
  red[tid] = m; __syncthreads();
  for (int st = 128; st > 0; st >>= 1) {
    if (tid < st) red[tid] = fmaxf(red[tid], red[tid + st]);
    __syncthreads();
  }
  m = red[0]; __syncthreads();
  float sum = 0.f;
  for (int i = 0; i < 4; ++i) { vals[i] = __expf(vals[i] - m); sum += vals[i]; }
  red[tid] = sum; __syncthreads();
  for (int st = 128; st > 0; st >>= 1) {
    if (tid < st) red[tid] += red[tid + st];
    __syncthreads();
  }
  const float rinv = 1.f / red[0];
  for (int i = 0; i < 4; ++i) row[tid + i * 256] = (_Float16)(vals[i] * rinv);
}

extern "C" void kernel_launch(void* const* d_in, const int* in_sizes, int n_in,
                              void* d_out, int out_size, void* d_ws, size_t ws_size,
                              hipStream_t stream)
{
  const float* x     = (const float*)d_in[0];
  const float* gamma = (const float*)d_in[1];
  const float* beta  = (const float*)d_in[2];
  const float* wq = (const float*)d_in[3];
  const float* bq = (const float*)d_in[4];
  const float* wk = (const float*)d_in[5];
  const float* bk = (const float*)d_in[6];
  const float* wv = (const float*)d_in[7];
  const float* bv = (const float*)d_in[8];
  const float* wp = (const float*)d_in[9];
  const float* bp = (const float*)d_in[10];
  float* out = (float*)d_out;

  char* ws = (char*)d_ws;
  size_t off = 0;
  auto take = [&](size_t bytes) {
    char* p = ws + off;
    off += (bytes + 255) & ~(size_t)255;
    return p;
  };
  _Float16* hnT = (_Float16*)take((size_t)BATCH * NTOK * CCH * 2);   // hn^T (b,n,c)
  _Float16* w16 = (_Float16*)take((size_t)4 * CCH * CCH * 2);        // wq|wk|wv|wp f16
  _Float16* qT  = (_Float16*)take((size_t)BATCH * NTOK * CCH * 2);   // q^T, later out_h^T
  _Float16* kT  = (_Float16*)take((size_t)BATCH * NTOK * CCH * 2);   // k^T
  _Float16* vb  = (_Float16*)take((size_t)BATCH * CCH * NTOK * 2);   // v (b,c,n)
  _Float16* sb  = (_Float16*)take((size_t)BATCH * NTOK * NTOK * 2);  // scores/attn

  _Float16* wq16 = w16 + (size_t)0 * CCH * CCH;
  _Float16* wk16 = w16 + (size_t)1 * CCH * CCH;
  _Float16* wv16 = w16 + (size_t)2 * CCH * CCH;
  _Float16* wp16 = w16 + (size_t)3 * CCH * CCH;

  const size_t sNC = (size_t)NTOK * CCH;
  const size_t sNN = (size_t)NTOK * NTOK;
  const size_t sCN = (size_t)CCH * NTOK;

  cvt_weights<<<4096, 256, 0, stream>>>(wq, wk, wv, wp, w16);
  groupnorm_to_hnT<<<dim3(NGRP, BATCH), 256, 0, stream>>>(x, gamma, beta, hnT);

  // q^T = (wq * hn)^T , k^T = (wk * hn)^T , v = wv * hn       (M=512, K=512)
  gemm16<1><<<dim3(8, 4, BATCH), 256, 0, stream>>>(wq16, 0, hnT, sNC, qT, nullptr, sNC,
                                                   bq, nullptr, 0, CCH, CCH, 1.f);
  gemm16<1><<<dim3(8, 4, BATCH), 256, 0, stream>>>(wk16, 0, hnT, sNC, kT, nullptr, sNC,
                                                   bk, nullptr, 0, CCH, CCH, 1.f);
  gemm16<2><<<dim3(8, 4, BATCH), 256, 0, stream>>>(wv16, 0, hnT, sNC, vb, nullptr, sNC,
                                                   bv, nullptr, 0, CCH, CCH, 1.f);

  // s = scale * q^T k   : A = q^T (n,c), B^T = k^T (m,c)      (M=1024, K=512)
  gemm16<0><<<dim3(8, 8, BATCH), 256, 0, stream>>>(qT, sNC, kT, sNC, sb, nullptr, sNN,
                                                   nullptr, nullptr, 0, NTOK, CCH,
                                                   0.044194173824159216f /* 512^-0.5 */);

  softmax_rows<<<dim3(NTOK, BATCH), 256, 0, stream>>>(sb);

  // out_h^T = (v * attn^T)^T : A = v (c,m), B^T = attn (n,m)  (M=512, K=1024)
  gemm16<1><<<dim3(8, 4, BATCH), 256, 0, stream>>>(vb, sCN, sb, sNN, qT, nullptr, sNC,
                                                   nullptr, nullptr, 0, CCH, NTOK, 1.f);

  // out = wp * out_h + bp + x  (f32, residual fused)          (M=512, K=512)
  gemm16<3><<<dim3(8, 4, BATCH), 256, 0, stream>>>(wp16, 0, qT, sNC, nullptr, out, sCN,
                                                   bp, x, sCN, CCH, CCH, 1.f);
}
